// MultiheadAttention_19825569038622
// MI455X (gfx1250) — matrix operand order repaired and checked by reading, not compiled, not run on hardware
//
#include <hip/hip_runtime.h>
#include <stdint.h>

#define D_MODEL 1024
#define N_HEADS 16
#define DH      64
#define SEQLEN  2048
#define BATCH   4
#define ROWS    (SEQLEN * BATCH)   // 8192 flattened (token, batch) rows

typedef __bf16 bf16_t;
typedef __attribute__((ext_vector_type(16))) __bf16 v16bf;
typedef __attribute__((ext_vector_type(8)))  __bf16 v8bf;
typedef __attribute__((ext_vector_type(4)))  __bf16 v4bf;
typedef __attribute__((ext_vector_type(8)))  float  v8f;
typedef __attribute__((ext_vector_type(4)))  float  v4f;
typedef unsigned int __attribute__((ext_vector_type(4))) u32x4;
typedef int  __attribute__((ext_vector_type(8))) i32x8;
typedef int  __attribute__((ext_vector_type(4))) i32x4;

#if __has_builtin(__builtin_amdgcn_tensor_load_to_lds)
#define HAVE_TDM 1
#endif

static __device__ __forceinline__ void wait_tensor0() {
#if __has_builtin(__builtin_amdgcn_s_wait_tensorcnt)
  __builtin_amdgcn_s_wait_tensorcnt(0);
#else
  asm volatile("s_wait_tensorcnt 0x0" ::: "memory");
#endif
}

#ifdef HAVE_TDM
// ---------------------------------------------------------------------------
// TDM 2D tile load: (tile_h x tile_w) bf16 tile from a row-major
// (tensor_h x tensor_w) tensor into LDS, with LDS padding of 4 DWORDs after
// every 16 DWORDs (64B row -> 80B stride, i.e. 40 bf16 elements per row).
// D# layout per CDNA5 ISA 8.3/8.4.
// ---------------------------------------------------------------------------
static __device__ __forceinline__ void tdm_load_tile_2d(uint32_t lds_addr,
                                                        const bf16_t* gsrc,
                                                        uint32_t tensor_w,
                                                        uint32_t tensor_h,
                                                        uint32_t tile_w,
                                                        uint32_t tile_h,
                                                        uint32_t row_stride) {
  const uint64_t ga = (uint64_t)(uintptr_t)gsrc;
  u32x4 g0;
  g0[0] = 1u;                                   // count=1, user descriptor
  g0[1] = lds_addr;                             // LDS byte address
  g0[2] = (uint32_t)ga;                         // global addr [31:0]
  g0[3] = (uint32_t)((ga >> 32) & 0x01FFFFFFu)  // global addr [56:32]
          | (2u << 30);                         // type = 2 ("image")
  i32x8 g1;
  g1[0] = (int)((1u << 16)                      // data_size = 2 bytes
              | (1u << 20)                      // pad_enable
              | (3u << 22)                      // pad_interval: 16 DWORDs
              | (3u << 25));                    // pad_amount: 4 DWORDs (16B)
  g1[1] = (int)((tensor_w & 0xFFFFu) << 16);            // tensor_dim0[15:0]
  g1[2] = (int)((tensor_w >> 16) | ((tensor_h & 0xFFFFu) << 16));
  g1[3] = (int)((tensor_h >> 16) | (tile_w << 16));     // tile_dim0
  g1[4] = (int)tile_h;                                  // tile_dim1 (tile_dim2=0)
  g1[5] = (int)row_stride;                              // tensor_dim0_stride[31:0]
  g1[6] = 0;
  g1[7] = 0;
  const i32x4 z4 = (i32x4)0;
#if __clang_major__ >= 23
  const i32x8 z8 = (i32x8)0;
  __builtin_amdgcn_tensor_load_to_lds(g0, g1, z4, z4, z8, 0);
#else
  __builtin_amdgcn_tensor_load_to_lds(g0, g1, z4, z4, 0);
#endif
}
#endif  // HAVE_TDM

// Cooperative fallback staging: 128x32 bf16 tile, 256 threads, stride-40 rows.
static __device__ __forceinline__ void stage_tile(bf16_t* __restrict__ dst,
                                                  const bf16_t* __restrict__ src,
                                                  size_t ld) {
  const int t = threadIdx.x;
  const int row = t >> 1, c = (t & 1) * 16;
  const bf16_t* s = src + (size_t)row * ld + c;
  v8bf a0 = *(const v8bf*)s;
  v8bf a1 = *(const v8bf*)(s + 8);
  *(v8bf*)(dst + row * 40 + c)     = a0;
  *(v8bf*)(dst + row * 40 + c + 8) = a1;
}

// ---------------------------------------------------------------------------
// fp32 -> bf16 elementwise conversion (vectorized), one-time pass.
// ---------------------------------------------------------------------------
__global__ __launch_bounds__(256)
void cvt_f32_to_bf16(const float* __restrict__ src, bf16_t* __restrict__ dst, int n4) {
  const int i = blockIdx.x * blockDim.x + threadIdx.x;
  if (i < n4) {
    v4f x = ((const v4f*)src)[i];
    v4bf y;
#pragma unroll
    for (int e = 0; e < 4; ++e) y[e] = (bf16_t)x[e];
    ((v4bf*)dst)[i] = y;
  }
}

// ---------------------------------------------------------------------------
// All-bf16 GEMM: Y = X @ W^T + bias.  Block = 256 threads (8 waves),
// tile 128(M) x 128(N), K-step 32.  X/W tiles staged in LDS (TDM when
// available), shared by all waves; inner loop is ds_read_b128 + v_wmma.
// Wave w owns rows [w*16, w*16+16) of the tile and all 128 columns.
// ---------------------------------------------------------------------------
template <bool OUTF32>
__global__ __launch_bounds__(256)
void gemm_tile128(const bf16_t* __restrict__ X, const bf16_t* __restrict__ W,
                  const float* __restrict__ bias, void* __restrict__ Yv) {
  __shared__ bf16_t Xs[128 * 40];   // 128 rows x 32 cols, 40-element stride
  __shared__ bf16_t Ws[128 * 40];

  const int wave = threadIdx.x >> 5;
  const int lane = threadIdx.x & 31;
  const int lr = lane & 15, hi = lane >> 4;
  const int m0 = blockIdx.x * 128;
  const int n0 = blockIdx.y * 128;

  v8f c[8];
#pragma unroll
  for (int f = 0; f < 8; ++f)
#pragma unroll
    for (int i = 0; i < 8; ++i) c[f][i] = 0.f;

  for (int k0 = 0; k0 < D_MODEL; k0 += 32) {
    __syncthreads();
#ifdef HAVE_TDM
    if (threadIdx.x < 32) {
      tdm_load_tile_2d((uint32_t)(uintptr_t)Xs, X + (size_t)m0 * D_MODEL + k0,
                       D_MODEL, ROWS, 32, 128, D_MODEL);
      tdm_load_tile_2d((uint32_t)(uintptr_t)Ws, W + (size_t)n0 * D_MODEL + k0,
                       D_MODEL, D_MODEL, 32, 128, D_MODEL);
      wait_tensor0();
    }
#else
    stage_tile(Xs, X + (size_t)m0 * D_MODEL + k0, D_MODEL);
    stage_tile(Ws, W + (size_t)n0 * D_MODEL + k0, D_MODEL);
#endif
    __syncthreads();

    // A fragment for this wave's 16 rows
    v16bf a;
    {
      const bf16_t* p = Xs + (wave * 16 + lr) * 40 + 8 * hi;
      v8bf x0 = *(const v8bf*)(p);
      v8bf x1 = *(const v8bf*)(p + 16);
#pragma unroll
      for (int e = 0; e < 8; ++e) { a[e] = x0[e]; a[8 + e] = x1[e]; }
    }
#pragma unroll
    for (int f = 0; f < 8; ++f) {
      const bf16_t* p = Ws + (f * 16 + lr) * 40 + 8 * hi;
      v8bf x0 = *(const v8bf*)(p);
      v8bf x1 = *(const v8bf*)(p + 16);
      v16bf w;
#pragma unroll
      for (int e = 0; e < 8; ++e) { w[e] = x0[e]; w[8 + e] = x1[e]; }
      c[f] = __builtin_amdgcn_wmma_f32_16x16x32_bf16(false, a, false, w,
                                                     (short)0, c[f], false, false);
    }
  }

#pragma unroll
  for (int f = 0; f < 8; ++f) {
    const int col = n0 + f * 16 + lr;
    const float bb = bias[col];
#pragma unroll
    for (int i = 0; i < 8; ++i) {
      const int row = m0 + wave * 16 + i + 8 * hi;
      if (OUTF32)
        ((float*)Yv)[(size_t)row * D_MODEL + col] = c[f][i] + bb;
      else
        ((bf16_t*)Yv)[(size_t)row * D_MODEL + col] = (bf16_t)(c[f][i] + bb);
    }
  }
}

// ---------------------------------------------------------------------------
// WMMA fragment helpers for flash attention (wave32, 16x16x32 bf16).
// ---------------------------------------------------------------------------
static __device__ __forceinline__ v16bf afrag_bf16(const bf16_t* __restrict__ p0,
                                                   size_t ld, int row0, int k0) {
  const int lane = threadIdx.x & 31;
  const int lr = lane & 15, hi = lane >> 4;
  const bf16_t* p = p0 + (size_t)(row0 + lr) * ld + (k0 + 8 * hi);
  v8bf x0 = *(const v8bf*)(p + 0);
  v8bf x1 = *(const v8bf*)(p + 16);
  v16bf a;
#pragma unroll
  for (int e = 0; e < 8; ++e) { a[e] = x0[e]; a[8 + e] = x1[e]; }
  return a;
}

static __device__ __forceinline__ v16bf bfrag_bf16(const bf16_t* __restrict__ p0,
                                                   size_t ld, int n0, int k0) {
  const int lane = threadIdx.x & 31;
  const int lr = lane & 15, hi = lane >> 4;
  const bf16_t* p = p0 + (size_t)(n0 + lr) * ld + (k0 + 8 * hi);
  v8bf x0 = *(const v8bf*)(p + 0);
  v8bf x1 = *(const v8bf*)(p + 16);
  v16bf b;
#pragma unroll
  for (int e = 0; e < 8; ++e) { b[e] = x0[e]; b[8 + e] = x1[e]; }
  return b;
}

// ---------------------------------------------------------------------------
// Flash attention: grid (L/64, B, H), 4 waves x 16 query rows, 32-key blocks,
// online softmax, causal mask.  P goes C-layout -> LDS(bf16) -> A-layout;
// V staged transposed in LDS so PV B-fragments are two ds_read_b128s.
// ---------------------------------------------------------------------------
__global__ __launch_bounds__(128)
void flash_attn_kernel(const bf16_t* __restrict__ Qb, const bf16_t* __restrict__ Kb,
                       const bf16_t* __restrict__ Vb, bf16_t* __restrict__ Ctx) {
  __shared__ bf16_t VT[DH][40];      // V^T tile [dh][key], padded row stride
  __shared__ bf16_t PT[4][16][32];   // per-wave probability tile

  const int tid  = threadIdx.x;
  const int wave = tid >> 5;
  const int lane = tid & 31;
  const int lr = lane & 15, hi = lane >> 4;

  const int q0 = blockIdx.x * 64;
  const int b  = blockIdx.y;
  const int h  = blockIdx.z;
  const int m_base = q0 + wave * 16;

  const size_t ld = (size_t)BATCH * D_MODEL;       // row stride between tokens
  const bf16_t* Qh = Qb + (size_t)b * D_MODEL + h * DH;
  const bf16_t* Kh = Kb + (size_t)b * D_MODEL + h * DH;
  const bf16_t* Vh = Vb + (size_t)b * D_MODEL + h * DH;

  const v16bf qa0 = afrag_bf16(Qh, ld, m_base, 0);
  const v16bf qa1 = afrag_bf16(Qh, ld, m_base, 32);

  v8f o[4];
  float mrun[8], lrun[8];
#pragma unroll
  for (int f = 0; f < 4; ++f)
#pragma unroll
    for (int i = 0; i < 8; ++i) o[f][i] = 0.f;
#pragma unroll
  for (int i = 0; i < 8; ++i) { mrun[i] = -1e30f; lrun[i] = 0.f; }

  const float scale = 0.03125f;      // D_MODEL^-0.5
  const int nblocks = q0 / 32 + 2;   // causal bound: keys <= q0+63

  for (int jb = 0; jb < nblocks; ++jb) {
    const int j0 = jb * 32;

    __syncthreads();
    { // cooperative transposed staging of the V tile (32 keys x 64 dh)
      const int key = tid & 31;
      const int dh0 = (tid >> 5) * 16;
      const bf16_t* src = Vh + (size_t)(j0 + key) * ld + dh0;
      v8bf v0 = *(const v8bf*)(src);
      v8bf v1 = *(const v8bf*)(src + 8);
#pragma unroll
      for (int e = 0; e < 8; ++e) {
        VT[dh0 + e][key]     = v0[e];
        VT[dh0 + 8 + e][key] = v1[e];
      }
    }
    __syncthreads();

    // scores S = Q K^T over 32 keys (two 16-key halves)
    v8f s0, s1;
#pragma unroll
    for (int i = 0; i < 8; ++i) { s0[i] = 0.f; s1[i] = 0.f; }
    {
      v16bf kb = bfrag_bf16(Kh, ld, j0, 0);
      s0 = __builtin_amdgcn_wmma_f32_16x16x32_bf16(false, qa0, false, kb, (short)0, s0, false, false);
      kb = bfrag_bf16(Kh, ld, j0, 32);
      s0 = __builtin_amdgcn_wmma_f32_16x16x32_bf16(false, qa1, false, kb, (short)0, s0, false, false);
      kb = bfrag_bf16(Kh, ld, j0 + 16, 0);
      s1 = __builtin_amdgcn_wmma_f32_16x16x32_bf16(false, qa0, false, kb, (short)0, s1, false, false);
      kb = bfrag_bf16(Kh, ld, j0 + 16, 32);
      s1 = __builtin_amdgcn_wmma_f32_16x16x32_bf16(false, qa1, false, kb, (short)0, s1, false, false);
    }

    // scale + causal mask (C layout: M = i + 8*hi, N = lr)
    float t[8];
#pragma unroll
    for (int i = 0; i < 8; ++i) {
      const int m = m_base + i + 8 * hi;
      s0[i] = (j0 + lr      <= m) ? s0[i] * scale : -__builtin_inff();
      s1[i] = (j0 + 16 + lr <= m) ? s1[i] * scale : -__builtin_inff();
      t[i] = fmaxf(s0[i], s1[i]);
    }
#pragma unroll
    for (int d = 1; d < 16; d <<= 1)
#pragma unroll
      for (int i = 0; i < 8; ++i) t[i] = fmaxf(t[i], __shfl_xor(t[i], d, 16));

    float p0[8], p1[8], r[8], alpha[8];
#pragma unroll
    for (int i = 0; i < 8; ++i) {
      const float mnew = fmaxf(mrun[i], t[i]);
      alpha[i] = __expf(mrun[i] - mnew);
      p0[i] = __expf(s0[i] - mnew);
      p1[i] = __expf(s1[i] - mnew);
      r[i] = p0[i] + p1[i];
      mrun[i] = mnew;
    }
#pragma unroll
    for (int d = 1; d < 16; d <<= 1)
#pragma unroll
      for (int i = 0; i < 8; ++i) r[i] += __shfl_xor(r[i], d, 16);
#pragma unroll
    for (int i = 0; i < 8; ++i) lrun[i] = lrun[i] * alpha[i] + r[i];
#pragma unroll
    for (int f = 0; f < 4; ++f)
#pragma unroll
      for (int i = 0; i < 8; ++i) o[f][i] *= alpha[i];

    // C-layout -> LDS -> A-layout conversion of P (wave-private tile)
#pragma unroll
    for (int i = 0; i < 8; ++i) {
      PT[wave][i + 8 * hi][lr]      = (bf16_t)p0[i];
      PT[wave][i + 8 * hi][16 + lr] = (bf16_t)p1[i];
    }
    asm volatile("s_wait_dscnt 0x0" ::: "memory");  // same-wave cross-lane LDS RAW

    v16bf pa;
    {
      const bf16_t* p = &PT[wave][lr][8 * hi];
      v8bf x0 = *(const v8bf*)(p);
      v8bf x1 = *(const v8bf*)(p + 16);
#pragma unroll
      for (int e = 0; e < 8; ++e) { pa[e] = x0[e]; pa[8 + e] = x1[e]; }
    }

    // O += P @ V  (4 dh column tiles, B-frags from transposed LDS tile)
#pragma unroll
    for (int f = 0; f < 4; ++f) {
      const bf16_t* p = &VT[f * 16 + lr][8 * hi];
      v8bf x0 = *(const v8bf*)(p);
      v8bf x1 = *(const v8bf*)(p + 16);
      v16bf vv;
#pragma unroll
      for (int e = 0; e < 8; ++e) { vv[e] = x0[e]; vv[8 + e] = x1[e]; }
      o[f] = __builtin_amdgcn_wmma_f32_16x16x32_bf16(false, pa, false, vv,
                                                     (short)0, o[f], false, false);
    }
  }

  // normalize and write context (same (L,B,D) flattening as inputs)
#pragma unroll
  for (int i = 0; i < 8; ++i) lrun[i] = 1.f / lrun[i];
#pragma unroll
  for (int f = 0; f < 4; ++f) {
    const int col = h * DH + f * 16 + lr;
#pragma unroll
    for (int i = 0; i < 8; ++i) {
      const int tok = m_base + i + 8 * hi;
      Ctx[((size_t)tok * BATCH + b) * D_MODEL + col] = (bf16_t)(o[f][i] * lrun[i]);
    }
  }
}

// ---------------------------------------------------------------------------
extern "C" void kernel_launch(void* const* d_in, const int* in_sizes, int n_in,
                              void* d_out, int out_size, void* d_ws, size_t ws_size,
                              hipStream_t stream) {
  (void)in_sizes; (void)n_in; (void)out_size; (void)ws_size;

  const float* q  = (const float*)d_in[0];
  const float* k  = (const float*)d_in[1];
  const float* v  = (const float*)d_in[2];
  // d_in[3] = attn_mask (bool causal) -- causality applied analytically
  const float* Wq = (const float*)d_in[4];
  const float* bq = (const float*)d_in[5];
  const float* Wk = (const float*)d_in[6];
  const float* bk = (const float*)d_in[7];
  const float* Wv = (const float*)d_in[8];
  const float* bv = (const float*)d_in[9];
  const float* Wo = (const float*)d_in[10];
  const float* bo = (const float*)d_in[11];

  const size_t n  = (size_t)ROWS * D_MODEL;      // activation elements
  const size_t nw = (size_t)D_MODEL * D_MODEL;   // weight elements
  bf16_t* qx = (bf16_t*)d_ws;        // bf16 activations
  bf16_t* kx = qx + n;
  bf16_t* vx = kx + n;
  bf16_t* Qb = vx + n;               // projected Q/K/V
  bf16_t* Kb = Qb + n;
  bf16_t* Vb = Kb + n;
  bf16_t* Cx = Vb + n;               // attention context
  bf16_t* wq = Cx + n;               // bf16 weights
  bf16_t* wk = wq + nw;
  bf16_t* wv = wk + nw;
  bf16_t* wo = wv + nw;

  // one-time fp32 -> bf16 conversion (bandwidth-bound)
  {
    const int na4 = (int)(n / 4), nw4 = (int)(nw / 4);
    dim3 cb(256);
    cvt_f32_to_bf16<<<dim3((na4 + 255) / 256), cb, 0, stream>>>(q,  qx, na4);
    cvt_f32_to_bf16<<<dim3((na4 + 255) / 256), cb, 0, stream>>>(k,  kx, na4);
    cvt_f32_to_bf16<<<dim3((na4 + 255) / 256), cb, 0, stream>>>(v,  vx, na4);
    cvt_f32_to_bf16<<<dim3((nw4 + 255) / 256), cb, 0, stream>>>(Wq, wq, nw4);
    cvt_f32_to_bf16<<<dim3((nw4 + 255) / 256), cb, 0, stream>>>(Wk, wk, nw4);
    cvt_f32_to_bf16<<<dim3((nw4 + 255) / 256), cb, 0, stream>>>(Wv, wv, nw4);
    cvt_f32_to_bf16<<<dim3((nw4 + 255) / 256), cb, 0, stream>>>(Wo, wo, nw4);
  }

  const dim3 gblk(256);
  const dim3 gg(ROWS / 128, D_MODEL / 128);
  gemm_tile128<false><<<gg, gblk, 0, stream>>>(qx, wq, bq, Qb);
  gemm_tile128<false><<<gg, gblk, 0, stream>>>(kx, wk, bk, Kb);
  gemm_tile128<false><<<gg, gblk, 0, stream>>>(vx, wv, bv, Vb);

  flash_attn_kernel<<<dim3(SEQLEN / 64, BATCH, N_HEADS), dim3(128), 0, stream>>>(Qb, Kb, Vb, Cx);

  gemm_tile128<true><<<gg, gblk, 0, stream>>>(Cx, wo, bo, (float*)d_out);
}
